// POCML_71803263254964
// MI455X (gfx1250) — matrix operand order repaired and checked by reading, not compiled, not run on hardware
//
#include <hip/hip_runtime.h>

// ---------------------------------------------------------------------------
// Problem constants (from reference)
// ---------------------------------------------------------------------------
#define SDIM      128     // STATE_DIM (K of the phase GEMM)
#define NS        4096    // N_STATES  (S)
#define NA        512     // N_ACTIONS
#define DD        16384   // D (random feature dim)
#define BB        16      // batch
#define INV2PI    0.15915494309189535f
#define SCALE_D   0.0078125f     // 1/sqrt(D) = 1/128
#define LOG2E     1.4426950408889634f

typedef __attribute__((ext_vector_type(16))) _Float16 v16h;
typedef __attribute__((ext_vector_type(8)))  float    v8f;
typedef __attribute__((ext_vector_type(4)))  unsigned v4u;

union Frag {
    v16h     v;
    v4u      q[2];
    _Float16 h[16];
};

static __device__ inline v8f zero8() {
    v8f z = {0.f,0.f,0.f,0.f,0.f,0.f,0.f,0.f};
    return z;
}

static __device__ inline v8f wmma_f16(v16h a, v16h b, v8f c) {
    // D = A(16x32 f16) * B(32x16 f16) + C(16x16 f32)
    return __builtin_amdgcn_wmma_f32_16x16x32_f16(
        /*neg_a=*/false, a, /*neg_b=*/false, b,
        /*c_mod=*/(short)0, c, /*reuse_a=*/false, /*reuse_b=*/false);
}

// A-fragment K offset, contiguous k order used by this code (not AMD's table order):
//  lanes 0-15 : elems 0..15 -> K 0..15
//  lanes 16-31: elems 0..15 -> K 16..31
static __device__ inline int kOffA(int lane, int e) {
    return ((lane < 16) ? 0 : 16) + e;
}

// ---------------------------------------------------------------------------
// Pack kernels: build WMMA fragments (f16) in workspace
// ---------------------------------------------------------------------------

// W [D,128] f32 row-major -> A-fragments. 1024 d-tiles x 4 k-chunks x 32 lanes.
__global__ __launch_bounds__(256) void pocml_pack_wA(const float* __restrict__ W,
                                                     v4u* __restrict__ dst) {
    int gid  = blockIdx.x * 256 + threadIdx.x;   // 131072 threads
    int lane = gid & 31;
    int kc   = (gid >> 5) & 3;
    int t    = gid >> 7;
    int d    = t * 16 + (lane & 15);
    Frag f;
#pragma unroll
    for (int e = 0; e < 16; ++e) {
        int k = kc * 32 + kOffA(lane, e);
        f.h[e] = (_Float16)W[d * SDIM + k];
    }
    dst[gid * 2 + 0] = f.q[0];
    dst[gid * 2 + 1] = f.q[1];
}

// Q [128,4096] f32 row-major -> B-fragments (32x16). 256 s-tiles x 4 kc x 32 lanes.
// B layout used here (contiguous k, matching kOffA): lane l holds col n=l%16; elems 0..15 -> K rows (l<16 ? 0..15 : 16..31).
__global__ __launch_bounds__(256) void pocml_pack_qB(const float* __restrict__ Q,
                                                     v4u* __restrict__ dst) {
    int gid  = blockIdx.x * 256 + threadIdx.x;   // 32768 threads
    int lane = gid & 31;
    int kc   = (gid >> 5) & 3;
    int u    = gid >> 7;
    int s    = u * 16 + (lane & 15);
    int kb   = kc * 32 + ((lane < 16) ? 0 : 16);
    Frag f;
#pragma unroll
    for (int e = 0; e < 16; ++e)
        f.h[e] = (_Float16)Q[(kb + e) * NS + s];
    dst[gid * 2 + 0] = f.q[0];
    dst[gid * 2 + 1] = f.q[1];
}

// obs [16,4096] f32 -> B-fragments of obs^T over S (128 chunks x 32 lanes).
__global__ __launch_bounds__(256) void pocml_pack_obsT(const float* __restrict__ obs,
                                                       v4u* __restrict__ dst) {
    int gid  = blockIdx.x * 256 + threadIdx.x;   // 4096 threads
    int lane = gid & 31;
    int c    = gid >> 5;
    int b    = lane & 15;
    int sb   = c * 32 + ((lane < 16) ? 0 : 16);
    Frag f;
#pragma unroll
    for (int e = 0; e < 16; ++e)
        f.h[e] = (_Float16)obs[b * NS + (sb + e)];
    dst[gid * 2 + 0] = f.q[0];
    dst[gid * 2 + 1] = f.q[1];
}

// Generic: U [K,16] f32 row-major -> B-fragments (K/32 chunks x 32 lanes).
__global__ __launch_bounds__(256) void pocml_pack_u(const float* __restrict__ U,
                                                    v4u* __restrict__ dst) {
    int gid  = blockIdx.x * 256 + threadIdx.x;
    int lane = gid & 31;
    int c    = gid >> 5;
    int n    = lane & 15;
    int kb   = c * 32 + ((lane < 16) ? 0 : 16);
    Frag f;
#pragma unroll
    for (int e = 0; e < 16; ++e)
        f.h[e] = (_Float16)U[(kb + e) * BB + n];
    dst[gid * 2 + 0] = f.q[0];
    dst[gid * 2 + 1] = f.q[1];
}

// ---------------------------------------------------------------------------
// Kernel A: Y_re/Y_im [D,16] = scale * (cosZ @ X , sinZ @ X), Z recomputed
// from fragments via WMMA. 1024 d-tiles = 128 blocks x 8 waves.
// ---------------------------------------------------------------------------
__global__ __launch_bounds__(256) void pocml_phiX(const v4u* __restrict__ wfrag,
                                                  const v4u* __restrict__ qfrag,
                                                  const v4u* __restrict__ xfrag,
                                                  float* __restrict__ yRe,
                                                  float* __restrict__ yIm,
                                                  float scale) {
    __shared__ float lcos[8][512];   // [d 16][s 32] per wave
    __shared__ float lsin[8][512];
    const int lane = threadIdx.x & 31;
    const int w    = threadIdx.x >> 5;
    const int t    = blockIdx.x * 8 + w;     // d-tile
    const int row  = lane & 15;
    const int hi8  = (lane < 16) ? 0 : 8;

    Frag wf[4];
#pragma unroll
    for (int kc = 0; kc < 4; ++kc) {
        const v4u* p = wfrag + ((size_t)(t * 4 + kc) * 32 + lane) * 2;
        wf[kc].q[0] = p[0];
        wf[kc].q[1] = p[1];
    }
    v8f aRe = zero8(), aIm = zero8();

    for (int sc = 0; sc < 128; ++sc) {       // 32 s-columns per chunk
#pragma unroll
        for (int h = 0; h < 2; ++h) {
            const int u = sc * 2 + h;
            v8f z = zero8();
#pragma unroll
            for (int kc = 0; kc < 4; ++kc) {
                Frag qf;
                const v4u* p = qfrag + ((size_t)(u * 4 + kc) * 32 + lane) * 2;
                qf.q[0] = p[0];
                qf.q[1] = p[1];
                z = wmma_f16(wf[kc].v, qf.v, z);   // Z tile of W@Q
            }
#pragma unroll
            for (int r = 0; r < 8; ++r) {
                const float zr = z[r] * INV2PI;    // v_cos/v_sin period is 1.0
                lcos[w][(r + hi8) * 32 + h * 16 + row] = __builtin_amdgcn_cosf(zr);
                lsin[w][(r + hi8) * 32 + h * 16 + row] = __builtin_amdgcn_sinf(zr);
            }
        }
        __syncthreads();
        Frag ca, sa, xb;
#pragma unroll
        for (int e = 0; e < 16; ++e) {
            const int k = kOffA(lane, e);
            ca.h[e] = (_Float16)lcos[w][row * 32 + k];
            sa.h[e] = (_Float16)lsin[w][row * 32 + k];
        }
        const v4u* p = xfrag + ((size_t)sc * 32 + lane) * 2;
        xb.q[0] = p[0];
        xb.q[1] = p[1];
        aRe = wmma_f16(ca.v, xb.v, aRe);
        aIm = wmma_f16(sa.v, xb.v, aIm);
        __syncthreads();
    }
#pragma unroll
    for (int r = 0; r < 8; ++r) {
        const int d = t * 16 + r + hi8;
        yRe[d * BB + row] = aRe[r] * scale;
        yIm[d * BB + row] = aIm[r] * scale;
    }
}

// ---------------------------------------------------------------------------
// Kernel B: R [S,16] = scale * (cosZ^T @ Ure + sinZ^T @ Uim).
// 256 s-tiles = 32 blocks x 8 waves; reduction over D in chunks of 32.
// ---------------------------------------------------------------------------
__global__ __launch_bounds__(256) void pocml_phiTU(const v4u* __restrict__ wfrag,
                                                   const v4u* __restrict__ qfrag,
                                                   const v4u* __restrict__ ureF,
                                                   const v4u* __restrict__ uimF,
                                                   float* __restrict__ out,
                                                   float scale) {
    __shared__ float lcos[8][512];   // [s 16][d 32] per wave (transposed stage)
    __shared__ float lsin[8][512];
    const int lane = threadIdx.x & 31;
    const int w    = threadIdx.x >> 5;
    const int u    = blockIdx.x * 8 + w;     // s-tile
    const int row  = lane & 15;
    const int hi8  = (lane < 16) ? 0 : 8;

    Frag qf[4];
#pragma unroll
    for (int kc = 0; kc < 4; ++kc) {
        const v4u* p = qfrag + ((size_t)(u * 4 + kc) * 32 + lane) * 2;
        qf[kc].q[0] = p[0];
        qf[kc].q[1] = p[1];
    }
    v8f acc = zero8();

    for (int dc = 0; dc < 512; ++dc) {       // 32 d-rows per chunk
#pragma unroll
        for (int h = 0; h < 2; ++h) {
            const int t = dc * 2 + h;
            v8f z = zero8();
#pragma unroll
            for (int kc = 0; kc < 4; ++kc) {
                Frag wfr;
                const v4u* p = wfrag + ((size_t)(t * 4 + kc) * 32 + lane) * 2;
                wfr.q[0] = p[0];
                wfr.q[1] = p[1];
                z = wmma_f16(wfr.v, qf[kc].v, z);
            }
#pragma unroll
            for (int r = 0; r < 8; ++r) {
                const float zr = z[r] * INV2PI;
                // transposed stage: [s_local][d_local]
                lcos[w][row * 32 + h * 16 + r + hi8] = __builtin_amdgcn_cosf(zr);
                lsin[w][row * 32 + h * 16 + r + hi8] = __builtin_amdgcn_sinf(zr);
            }
        }
        __syncthreads();
        Frag ca, sa, ur, ui;
#pragma unroll
        for (int e = 0; e < 16; ++e) {
            const int k = kOffA(lane, e);
            ca.h[e] = (_Float16)lcos[w][row * 32 + k];
            sa.h[e] = (_Float16)lsin[w][row * 32 + k];
        }
        const v4u* pr = ureF + ((size_t)dc * 32 + lane) * 2;
        const v4u* pi = uimF + ((size_t)dc * 32 + lane) * 2;
        ur.q[0] = pr[0]; ur.q[1] = pr[1];
        ui.q[0] = pi[0]; ui.q[1] = pi[1];
        acc = wmma_f16(ca.v, ur.v, acc);
        acc = wmma_f16(sa.v, ui.v, acc);
        __syncthreads();
    }
#pragma unroll
    for (int r = 0; r < 8; ++r) {
        const int s = u * 16 + r + hi8;
        out[s * BB + row] = acc[r] * scale;
    }
}

// ---------------------------------------------------------------------------
// Softmax over S (axis 0) per batch column. 16 blocks x 256 threads.
// ---------------------------------------------------------------------------
__global__ __launch_bounds__(256) void pocml_softmax(const float* __restrict__ in,
                                                     float* __restrict__ out) {
    __shared__ float red[256];
    const int b   = blockIdx.x;
    const int tid = threadIdx.x;

    float m = -3.4e38f;
    for (int s = tid; s < NS; s += 256) m = fmaxf(m, in[s * BB + b]);
    red[tid] = m;
    __syncthreads();
    for (int o = 128; o > 0; o >>= 1) {
        if (tid < o) red[tid] = fmaxf(red[tid], red[tid + o]);
        __syncthreads();
    }
    m = red[0];
    __syncthreads();

    float sum = 0.f;
    for (int s = tid; s < NS; s += 256) {
        float e = __builtin_amdgcn_exp2f((in[s * BB + b] - m) * LOG2E);
        out[s * BB + b] = e;
        sum += e;
    }
    red[tid] = sum;
    __syncthreads();
    for (int o = 128; o > 0; o >>= 1) {
        if (tid < o) red[tid] += red[tid + o];
        __syncthreads();
    }
    const float inv = 1.f / red[0];
    for (int s = tid; s < NS; s += 256) out[s * BB + b] *= inv;
}

// ---------------------------------------------------------------------------
// v = V @ action^T : [128,16]. 8 blocks x 256 threads.
// ---------------------------------------------------------------------------
__global__ __launch_bounds__(256) void pocml_vmat(const float* __restrict__ V,
                                                  const float* __restrict__ action,
                                                  float* __restrict__ v) {
    int tid = blockIdx.x * 256 + threadIdx.x;   // 2048
    int i = tid >> 4, b = tid & 15;
    float acc = 0.f;
    for (int a = 0; a < NA; ++a) acc += V[i * NA + a] * action[b * NA + a];
    v[i * BB + b] = acc;
}

// ---------------------------------------------------------------------------
// out_state = new_state * phi(W@v): zv per (d,b), complex multiply.
// 1024 blocks x 256 threads (one thread per (d,b)).
// ---------------------------------------------------------------------------
__global__ __launch_bounds__(256) void pocml_outstate(const float* __restrict__ W,
                                                      const float* __restrict__ v,
                                                      const float* __restrict__ nsRe,
                                                      const float* __restrict__ nsIm,
                                                      float* __restrict__ oRe,
                                                      float* __restrict__ oIm) {
    __shared__ float vs[SDIM * BB];
    for (int idx = threadIdx.x; idx < SDIM * BB; idx += 256) vs[idx] = v[idx];
    __syncthreads();
    int tid = blockIdx.x * 256 + threadIdx.x;
    int d = tid >> 4, b = tid & 15;
    float zv = 0.f;
    for (int i = 0; i < SDIM; ++i) zv += W[d * SDIM + i] * vs[i * BB + b];
    zv *= INV2PI;
    float c = __builtin_amdgcn_cosf(zv) * SCALE_D;
    float s = __builtin_amdgcn_sinf(zv) * SCALE_D;
    float nr = nsRe[tid], ni = nsIm[tid];
    oRe[tid] = nr * c - ni * s;
    oIm[tid] = nr * s + ni * c;
}

// ---------------------------------------------------------------------------
// Host-side orchestration
// ---------------------------------------------------------------------------
extern "C" void kernel_launch(void* const* d_in, const int* in_sizes, int n_in,
                              void* d_out, int out_size, void* d_ws, size_t ws_size,
                              hipStream_t stream) {
    (void)in_sizes; (void)n_in; (void)out_size; (void)ws_size;
    const float* obs    = (const float*)d_in[0];   // [16,4096]
    const float* action = (const float*)d_in[1];   // [16,512]
    const float* Q      = (const float*)d_in[2];   // [128,4096]
    const float* V      = (const float*)d_in[3];   // [128,512]
    const float* W      = (const float*)d_in[4];   // [16384,128]
    float* out = (float*)d_out;                    // [4096,16]

    char*  ws  = (char*)d_ws;
    size_t off = 0;
    auto carve = [&](size_t bytes) -> void* {
        void* p = ws + off;
        off += (bytes + 255) & ~(size_t)255;
        return p;
    };

    v4u*   wfrag  = (v4u*)carve((size_t)1024 * 4 * 32 * 32);  // 4 MB  W A-fragments
    v4u*   qfrag  = (v4u*)carve((size_t)256  * 4 * 32 * 32);  // 1 MB  Q B-fragments
    v4u*   xobs   = (v4u*)carve((size_t)128  * 32 * 32);      // obs^T B-fragments
    float* stRe   = (float*)carve((size_t)DD * BB * 4);
    float* stIm   = (float*)carve((size_t)DD * BB * 4);
    v4u*   stFr   = (v4u*)carve((size_t)512 * 32 * 32);
    v4u*   stFi   = (v4u*)carve((size_t)512 * 32 * 32);
    float* scores = (float*)carve((size_t)NS * BB * 4);
    float* wts    = (float*)carve((size_t)NS * BB * 4);
    v4u*   wxf    = (v4u*)carve((size_t)128 * 32 * 32);
    float* nsRe   = (float*)carve((size_t)DD * BB * 4);
    float* nsIm   = (float*)carve((size_t)DD * BB * 4);
    float* vbuf   = (float*)carve((size_t)SDIM * BB * 4);
    float* oRe    = (float*)carve((size_t)DD * BB * 4);
    float* oIm    = (float*)carve((size_t)DD * BB * 4);
    v4u*   oFr    = (v4u*)carve((size_t)512 * 32 * 32);
    v4u*   oFi    = (v4u*)carve((size_t)512 * 32 * 32);
    float* pres   = (float*)carve((size_t)NS * BB * 4);

    // 0) pack W / Q / obs^T into WMMA fragments (L2-resident thereafter)
    pocml_pack_wA<<<512, 256, 0, stream>>>(W, wfrag);
    pocml_pack_qB<<<128, 256, 0, stream>>>(Q, qfrag);
    pocml_pack_obsT<<<16, 256, 0, stream>>>(obs, xobs);

    // 1) state = phi_Q @ obs^T
    pocml_phiX<<<128, 256, 0, stream>>>(wfrag, qfrag, xobs, stRe, stIm, SCALE_D);

    // 2) scores = beta * Re(phi_Q^H @ state)
    pocml_pack_u<<<64, 256, 0, stream>>>(stRe, stFr);
    pocml_pack_u<<<64, 256, 0, stream>>>(stIm, stFi);
    pocml_phiTU<<<32, 256, 0, stream>>>(wfrag, qfrag, stFr, stFi, scores, SCALE_D);

    // 3) weights = softmax(scores, axis=0); new_state = phi_Q @ weights
    pocml_softmax<<<16, 256, 0, stream>>>(scores, wts);
    pocml_pack_u<<<16, 256, 0, stream>>>(wts, wxf);
    pocml_phiX<<<128, 256, 0, stream>>>(wfrag, qfrag, wxf, nsRe, nsIm, SCALE_D);

    // 4) out_state = new_state * phi(W @ (V @ action^T))
    pocml_vmat<<<8, 256, 0, stream>>>(V, action, vbuf);
    pocml_outstate<<<1024, 256, 0, stream>>>(W, vbuf, nsRe, nsIm, oRe, oIm);

    // 5) out = softmax(beta * Re(phi_Q^H @ out_state), axis=0)
    pocml_pack_u<<<64, 256, 0, stream>>>(oRe, oFr);
    pocml_pack_u<<<64, 256, 0, stream>>>(oIm, oFi);
    pocml_phiTU<<<32, 256, 0, stream>>>(wfrag, qfrag, oFr, oFi, pres, SCALE_D);
    pocml_softmax<<<16, 256, 0, stream>>>(pres, out);
}
